// AdaptiveAttentionKernel_55851754717316
// MI455X (gfx1250) — matrix operand order repaired and checked by reading, not compiled, not run on hardware
//
#include <hip/hip_runtime.h>
#include <hip/hip_bf16.h>
#include <math.h>

typedef _Float16 half_t;
typedef __attribute__((ext_vector_type(16))) _Float16 v16h;
typedef __attribute__((ext_vector_type(8)))  _Float16 v8h;
typedef __attribute__((ext_vector_type(8)))  float    v8f;

#define DEV static __device__ __forceinline__

// ---------------- WMMA helpers (gfx1250, wave32) ----------------

DEV v8f wmma_f16(v16h a, v16h b, v8f c) {
  // D = A(16x32 f16) * B(32x16 f16) + C(16x16 f32)
  return __builtin_amdgcn_wmma_f32_16x16x32_f16(false, a, false, b, (short)0, c,
                                                false, false);
}

DEV v8f zero8() {
  v8f z = {0.f, 0.f, 0.f, 0.f, 0.f, 0.f, 0.f, 0.f};
  return z;
}

// A fragment 16x32, row-major f16 source, base already at (m0,k0).
// ISA layout: lanes 0-15 -> M=lane, K={0..7,16..23}; lanes 16-31 -> K={8..15,24..31}
DEV v16h load_a_g(const half_t* __restrict__ base, int lda, int lane) {
  const int m  = lane & 15;
  const int kb = (lane >> 4) << 3;   // 0 or 8
  const half_t* p = base + (long)m * lda + kb;
  v8h lo = *(const v8h*)(p);
  v8h hi = *(const v8h*)(p + 16);
  v16h r;
#pragma unroll
  for (int i = 0; i < 8; ++i) { r[i] = lo[i]; r[i + 8] = hi[i]; }
  return r;
}

// B fragment 32x16, row-major [K,N] f16 source, base already at (k0,n0).
// ISA layout: lanes 0-15 -> N=lane, K={0..7,16..23}; lanes 16-31 -> K={8..15,24..31}
DEV v16h load_b_g(const half_t* __restrict__ base, int ldb, int lane) {
  const int n  = lane & 15;
  const int kb = (lane >> 4) << 3;
  const half_t* p = base + (long)kb * ldb + n;
  v16h r;
#pragma unroll
  for (int i = 0; i < 8; ++i) { r[i] = p[(long)i * ldb]; r[i + 8] = p[(long)(i + 16) * ldb]; }
  return r;
}

// ---------------- generic WMMA GEMM: C = A[M,K] * B[K,N] + bias ----------------
// block = 256 threads = 8 waves (4 m-tiles x 2 n-tiles); wave tile = 16 x (NFRAG*16).
// NO bounds guards: launch geometry must tile M,N exactly (all shapes here do).
// gridDim.z = heads, with per-head element strides a_hs/b_hs/c_hs.

template <bool OUT_HALF, int NFRAG>
__global__ void __launch_bounds__(256)
gemm_kernel(const half_t* __restrict__ A, int lda, long a_hs,
            const half_t* __restrict__ B, int ldb, long b_hs,
            const float* __restrict__ bias,
            void* __restrict__ Cp, int ldc, long c_hs, int K) {
  const int lane = threadIdx.x & 31;
  const int wid  = threadIdx.x >> 5;
  const int wm   = wid & 3, wn = wid >> 2;
  const int m0   = blockIdx.x * 64 + wm * 16;
  const int n0   = blockIdx.y * (NFRAG * 32) + wn * (NFRAG * 16);
  const int h    = blockIdx.z;
  A += (long)h * a_hs + (long)m0 * lda;
  B += (long)h * b_hs + n0;

  v8f acc[NFRAG];
#pragma unroll
  for (int f = 0; f < NFRAG; ++f) acc[f] = zero8();

  for (int k0 = 0; k0 < K; k0 += 32) {
    v16h a = load_a_g(A + k0, lda, lane);
#pragma unroll
    for (int f = 0; f < NFRAG; ++f) {
      v16h b = load_b_g(B + (long)k0 * ldb + f * 16, ldb, lane);
      acc[f] = wmma_f16(a, b, acc[f]);
    }
  }

  const int cn = lane & 15, chi = lane >> 4;
#pragma unroll
  for (int f = 0; f < NFRAG; ++f) {
    int n = n0 + f * 16 + cn;
    float bv = bias ? bias[n] : 0.f;
#pragma unroll
    for (int r = 0; r < 8; ++r) {
      int m = m0 + r + (chi << 3);
      float v = acc[f][r] + bv;
      if (OUT_HALF)
        ((half_t*)Cp)[(long)h * c_hs + (long)m * ldc + n] = (half_t)v;
      else
        ((float*)Cp)[(long)h * c_hs + (long)m * ldc + n] = v;
    }
  }
}

// ---------------- per-head coefficient on scores ----------------
DEV float score_coef(int mode, const float* __restrict__ src, int h) {
  if (mode == 1) return (src[h] > 0.f) ? 1.f : 0.f;            // sigmoid(x)>0.5
  if (mode == 2) return 1.f / (1.f + __expf(-src[h * 4]));     // sigmoid(ent_gate[h,0,0])
  return 1.f;
}

// ---------------- attention pass 1: per-row softmax stats (m, l) ----------------
// one wave per (head, 16-row tile); scores via WMMA; cross-lane max/sum.
// grid is sized EXACTLY: blocks = Hn*(Tq/16)/8 (always integral here) -> no guards.

template <int HDv>
__global__ void __launch_bounds__(256)
attn_stats(const half_t* __restrict__ Q, int ldq,
           const half_t* __restrict__ KT,          // [H][HDv][Tk] f16
           float* __restrict__ stats_m, float* __restrict__ stats_l,
           int Hn, int Tq, int Tk, float inv_scale,
           const float* __restrict__ coef_src, int coef_mode) {
  const int lane = threadIdx.x & 31;
  const int wid  = threadIdx.x >> 5;
  const int gw   = blockIdx.x * 8 + wid;
  const int nt   = Tq / 16;
  const int tt   = gw % nt;
  const int h    = gw / nt;
  const int t0   = tt * 16;
  const float sc = inv_scale * score_coef(coef_mode, coef_src, h);

  v16h aq[HDv / 32];
#pragma unroll
  for (int ka = 0; ka < HDv / 32; ++ka)
    aq[ka] = load_a_g(Q + (long)t0 * ldq + h * HDv + ka * 32, ldq, lane);

  float m[8], l[8];
#pragma unroll
  for (int r = 0; r < 8; ++r) { m[r] = -1e30f; l[r] = 0.f; }

  const half_t* KTh = KT + (long)h * HDv * Tk;
  for (int s0 = 0; s0 < Tk; s0 += 32) {
    v8f s0f = zero8(), s1f = zero8();
#pragma unroll
    for (int ka = 0; ka < HDv / 32; ++ka) {
      v16h b0 = load_b_g(KTh + (long)(ka * 32) * Tk + s0, Tk, lane);
      v16h b1 = load_b_g(KTh + (long)(ka * 32) * Tk + s0 + 16, Tk, lane);
      s0f = wmma_f16(aq[ka], b0, s0f);
      s1f = wmma_f16(aq[ka], b1, s1f);
    }
#pragma unroll
    for (int r = 0; r < 8; ++r) {
      float v0 = s0f[r] * sc, v1 = s1f[r] * sc;
      float tm = fmaxf(v0, v1);
#pragma unroll
      for (int msk = 1; msk <= 8; msk <<= 1) tm = fmaxf(tm, __shfl_xor(tm, msk, 32));
      float mn = fmaxf(m[r], tm);
      float e  = __expf(v0 - mn) + __expf(v1 - mn);
#pragma unroll
      for (int msk = 1; msk <= 8; msk <<= 1) e += __shfl_xor(e, msk, 32);
      l[r] = l[r] * __expf(m[r] - mn) + e;
      m[r] = mn;
    }
  }
  if ((lane & 15) == 0) {
    const int hi = lane >> 4;
#pragma unroll
    for (int r = 0; r < 8; ++r) {
      int row = t0 + r + (hi << 3);
      stats_m[(long)h * Tq + row] = m[r];
      stats_l[(long)h * Tq + row] = l[r];
    }
  }
}

// ---------------- attention pass 2: probs, combined-accum, p@V ----------------

template <int HDv>
__global__ void __launch_bounds__(256)
attn_apply(const half_t* __restrict__ Q, int ldq,
           const half_t* __restrict__ KT,
           const half_t* __restrict__ V, int ldv,
           const float* __restrict__ stats_m, const float* __restrict__ stats_l,
           float* __restrict__ Out, int ldo,
           int Hn, int Tq, int Tk, float inv_scale,
           const float* __restrict__ coef_src, int coef_mode,
           float* __restrict__ comb, const float* __restrict__ sprobs, int sp_idx) {
  __shared__ __align__(16) half_t pbuf[8][16 * 32];
  const int lane = threadIdx.x & 31;
  const int wid  = threadIdx.x >> 5;
  const int gw   = blockIdx.x * 8 + wid;
  const int nt   = Tq / 16;
  const int tt   = gw % nt;
  const int h    = gw / nt;
  const int t0   = tt * 16;
  const float sc = inv_scale * score_coef(coef_mode, coef_src, h);
  const bool  do_comb = (comb != nullptr);
  const float cw = do_comb ? (sprobs[sp_idx] / (float)Hn) : 0.f;

  v16h aq[HDv / 32];
#pragma unroll
  for (int ka = 0; ka < HDv / 32; ++ka)
    aq[ka] = load_a_g(Q + (long)t0 * ldq + h * HDv + ka * 32, ldq, lane);

  const int cn = lane & 15, chi = lane >> 4;
  float mrow[8], lrow[8];
#pragma unroll
  for (int r = 0; r < 8; ++r) {
    int row = t0 + r + (chi << 3);
    mrow[r] = stats_m[(long)h * Tq + row];
    lrow[r] = 1.f / stats_l[(long)h * Tq + row];
  }

  v8f o[HDv / 16];
#pragma unroll
  for (int f = 0; f < HDv / 16; ++f) o[f] = zero8();

  half_t* myp = &pbuf[wid][0];
  const half_t* KTh = KT + (long)h * HDv * Tk;
  const half_t* Vh  = V + h * HDv;

  for (int s0 = 0; s0 < Tk; s0 += 32) {
#pragma unroll
    for (int j = 0; j < 2; ++j) {
      v8f s = zero8();
#pragma unroll
      for (int ka = 0; ka < HDv / 32; ++ka) {
        v16h b = load_b_g(KTh + (long)(ka * 32) * Tk + s0 + j * 16, Tk, lane);
        s = wmma_f16(aq[ka], b, s);
      }
#pragma unroll
      for (int r = 0; r < 8; ++r) {
        float p = __expf(s[r] * sc - mrow[r]) * lrow[r];
        if (do_comb)
          atomicAdd(&comb[(long)(t0 + r + (chi << 3)) * Tk + s0 + j * 16 + cn], cw * p);
        myp[(r + (chi << 3)) * 32 + j * 16 + cn] = (half_t)p;
      }
    }
    // p (16x32) @ V (32xHDv)  -- LDS round-trip converts C-layout -> A-layout
    v16h pa = load_a_g(myp, 32, lane);
#pragma unroll
    for (int f = 0; f < HDv / 16; ++f) {
      v16h bv = load_b_g(Vh + (long)s0 * ldv + f * 16, ldv, lane);
      o[f] = wmma_f16(pa, bv, o[f]);
    }
  }

#pragma unroll
  for (int f = 0; f < HDv / 16; ++f)
#pragma unroll
    for (int r = 0; r < 8; ++r)
      Out[(long)(t0 + r + (chi << 3)) * ldo + h * HDv + f * 16 + cn] = o[f][r];
}

// ---------------- small elementwise / layout kernels ----------------

__global__ void zero_f32(float* p, long n) {
  long i = (long)blockIdx.x * blockDim.x + threadIdx.x;
  if (i < n) p[i] = 0.f;
}

__global__ void f32_to_f16(const float* __restrict__ s, half_t* __restrict__ d, long n) {
  long i = (long)blockIdx.x * blockDim.x + threadIdx.x;
  if (i < n) d[i] = (half_t)s[i];
}

// W [N,K] f32 -> Wt [K,N] f16
__global__ void transpose_w(const float* __restrict__ W, half_t* __restrict__ Wt,
                            int N, int K) {
  long i = (long)blockIdx.x * blockDim.x + threadIdx.x;
  long total = (long)N * K;
  if (i < total) {
    int n = (int)(i % N);
    int k = (int)(i / N);
    Wt[(long)k * N + n] = (half_t)W[(long)n * K + k];
  }
}

// KT[h][e][t] = src[t*ld + col0 + h*hd + e]
__global__ void head_transpose(const half_t* __restrict__ src, int ld, int col0,
                               half_t* __restrict__ dst, int H, int HDv, int T) {
  long i = (long)blockIdx.x * blockDim.x + threadIdx.x;
  long total = (long)H * HDv * T;
  if (i < total) {
    int t = (int)(i % T);
    long r = i / T;
    int e = (int)(r % HDv);
    int h = (int)(r / HDv);
    dst[i] = src[(long)t * ld + col0 + h * HDv + e];
  }
}

__global__ void sample_f16(const float* __restrict__ src, half_t* __restrict__ dst,
                           int Ts, int C, int stride) {
  long i = (long)blockIdx.x * blockDim.x + threadIdx.x;
  if (i < (long)Ts * C) {
    int t = (int)(i / C), c = (int)(i % C);
    dst[i] = (half_t)src[(long)t * stride * C + c];
  }
}

// concat[t, 0:D] = local_out; concat[t, D:2D] = linear-interp(g_small) (f16)
__global__ void concat_interp(const float* __restrict__ loc,
                              const float* __restrict__ gsm,
                              half_t* __restrict__ cat, int T, int D, int Ts) {
  long i = (long)blockIdx.x * blockDim.x + threadIdx.x;
  if (i < (long)T * D) {
    int t = (int)(i / D), c = (int)(i % D);
    cat[(long)t * 2 * D + c] = (half_t)loc[i];
    float src = (t + 0.5f) * ((float)Ts / (float)T) - 0.5f;
    src = fminf(fmaxf(src, 0.f), (float)Ts - 1.f);
    int i0 = (int)floorf(src);
    int i1 = min(i0 + 1, Ts - 1);
    float w = src - (float)i0;
    float g = gsm[(long)i0 * D + c] * (1.f - w) + gsm[(long)i1 * D + c] * w;
    cat[(long)t * 2 * D + D + c] = (half_t)g;
  }
}

__global__ void final_combine(const float* __restrict__ lin, const float* __restrict__ spo,
                              const float* __restrict__ hier, const float* __restrict__ quo,
                              const float* __restrict__ sp, float* __restrict__ out, long n) {
  long i = (long)blockIdx.x * blockDim.x + threadIdx.x;
  if (i < n)
    out[i] = sp[0] * lin[i] + sp[1] * spo[i] + sp[2] * hier[i] + sp[3] * quo[i];
}

// ---------------- host orchestration ----------------

static inline dim3 ew_grid(long n) { return dim3((unsigned)((n + 255) / 256)); }

extern "C" void kernel_launch(void* const* d_in, const int* in_sizes, int n_in,
                              void* d_out, int out_size, void* d_ws, size_t ws_size,
                              hipStream_t stream) {
  (void)in_sizes; (void)n_in; (void)out_size; (void)ws_size;
  const int T = 2048, D = 1024, H = 16, HD = 64, TG = 256, D3 = 3072, D2 = 2048;
  const int H2 = 8, HD2 = 128;
  const float ISC64  = 0.125f;                 // 1/sqrt(64)
  const float ISC128 = 0.0883883476f;          // 1/sqrt(128)

  const float* query   = (const float*)d_in[0];
  const float* key     = (const float*)d_in[1];
  const float* value   = (const float*)d_in[2];
  const float* sprobs  = (const float*)d_in[3];
  const float* Wqkv_l  = (const float*)d_in[4];
  const float* b_l     = (const float*)d_in[5];
  const float* Wo_l    = (const float*)d_in[6];
  const float* bo_l    = (const float*)d_in[7];
  const float* spat    = (const float*)d_in[8];
  const float* Wqkv_lo = (const float*)d_in[9];
  const float* b_lo    = (const float*)d_in[10];
  const float* Wo_lo   = (const float*)d_in[11];
  const float* bo_lo   = (const float*)d_in[12];
  const float* Wqkv_g  = (const float*)d_in[13];
  const float* b_g     = (const float*)d_in[14];
  const float* Wo_g    = (const float*)d_in[15];
  const float* bo_g    = (const float*)d_in[16];
  const float* Wf      = (const float*)d_in[17];
  const float* bf      = (const float*)d_in[18];
  const float* Rq      = (const float*)d_in[19];
  const float* entg    = (const float*)d_in[20];

  float* weighted = (float*)d_out;
  float* combined = (float*)d_out + (long)T * D;

  // bump allocator over d_ws
  char* wsb = (char*)d_ws;
  size_t off = 0;
  auto alloc = [&](size_t bytes) -> void* {
    void* p = wsb + off;
    off = (off + bytes + 255) & ~(size_t)255;
    return p;
  };
  auto hbuf = [&](long n) { return (half_t*)alloc((size_t)n * sizeof(half_t)); };
  auto fbuf = [&](long n) { return (float*)alloc((size_t)n * sizeof(float)); };

  half_t* qf = hbuf((long)T * D);
  half_t* kf = hbuf((long)T * D);
  half_t* vf = hbuf((long)T * D);
  half_t* WqkvT_l  = hbuf((long)D * D3);
  half_t* WqkvT_lo = hbuf((long)D * D3);
  half_t* WqkvT_g  = hbuf((long)D * D3);
  half_t* WoT_l  = hbuf((long)D * D);
  half_t* WoT_lo = hbuf((long)D * D);
  half_t* WoT_g  = hbuf((long)D * D);
  half_t* WfT    = hbuf((long)D2 * D);
  half_t* Rf     = hbuf((long)H * HD * HD);
  half_t* Qp_l = hbuf((long)T * D);
  half_t* Kp_l = hbuf((long)T * D);
  half_t* Vp_l = hbuf((long)T * D);
  half_t* Qp_lo = hbuf((long)T * D);
  half_t* Kp_lo = hbuf((long)T * D);
  half_t* Vp_lo = hbuf((long)T * D);
  half_t* qs = hbuf((long)TG * D);
  half_t* ks = hbuf((long)TG * D);
  half_t* vs = hbuf((long)TG * D);
  half_t* Qp_g = hbuf((long)TG * D);
  half_t* Kp_g = hbuf((long)TG * D);
  half_t* Vp_g = hbuf((long)TG * D);
  half_t* KT_l  = hbuf((long)H * HD * T);
  half_t* KT_sp = hbuf((long)H * HD * T);
  half_t* KT_qr = hbuf((long)H * HD * T);
  half_t* KT_lo = hbuf((long)H2 * HD2 * T);
  half_t* KT_g  = hbuf((long)H2 * HD2 * TG);
  half_t* qrot = hbuf((long)T * D);
  half_t* krot = hbuf((long)T * D);
  float*  at_l_f  = fbuf((long)T * D);
  half_t* at_l_h  = hbuf((long)T * D);
  float*  at_lo_f = fbuf((long)T * D);
  half_t* at_lo_h = hbuf((long)T * D);
  float*  at_g_f  = fbuf((long)TG * D);
  half_t* at_g_h  = hbuf((long)TG * D);
  float*  sparse_o = fbuf((long)T * D);
  float*  quant_o  = fbuf((long)T * D);
  float*  lin_o    = fbuf((long)T * D);
  float*  loc_o    = fbuf((long)T * D);
  float*  gsm      = fbuf((long)TG * D);
  half_t* catb     = hbuf((long)T * D2);
  float*  hier_o   = fbuf((long)T * D);
  float* sm1 = fbuf((long)H * T); float* sl1 = fbuf((long)H * T);
  float* sm2 = fbuf((long)H * T); float* sl2 = fbuf((long)H * T);
  float* sm3 = fbuf((long)H * T); float* sl3 = fbuf((long)H * T);
  float* sm4 = fbuf((long)H * T); float* sl4 = fbuf((long)H * T);
  float* sm5 = fbuf((long)H * T); float* sl5 = fbuf((long)H * T);

  const dim3 B256(256);

  // 0) zero the combined output region (atomically accumulated later)
  zero_f32<<<ew_grid((long)T * T), B256, 0, stream>>>(combined, (long)T * T);

  // 1) precision conversions / weight transposes
  f32_to_f16<<<ew_grid((long)T * D), B256, 0, stream>>>(query, qf, (long)T * D);
  f32_to_f16<<<ew_grid((long)T * D), B256, 0, stream>>>(key,   kf, (long)T * D);
  f32_to_f16<<<ew_grid((long)T * D), B256, 0, stream>>>(value, vf, (long)T * D);
  transpose_w<<<ew_grid((long)D3 * D), B256, 0, stream>>>(Wqkv_l,  WqkvT_l,  D3, D);
  transpose_w<<<ew_grid((long)D3 * D), B256, 0, stream>>>(Wqkv_lo, WqkvT_lo, D3, D);
  transpose_w<<<ew_grid((long)D3 * D), B256, 0, stream>>>(Wqkv_g,  WqkvT_g,  D3, D);
  transpose_w<<<ew_grid((long)D * D),  B256, 0, stream>>>(Wo_l,  WoT_l,  D, D);
  transpose_w<<<ew_grid((long)D * D),  B256, 0, stream>>>(Wo_lo, WoT_lo, D, D);
  transpose_w<<<ew_grid((long)D * D),  B256, 0, stream>>>(Wo_g,  WoT_g,  D, D);
  transpose_w<<<ew_grid((long)D * D2), B256, 0, stream>>>(Wf, WfT, D, D2);
  f32_to_f16<<<ew_grid((long)H * HD * HD), B256, 0, stream>>>(Rq, Rf, (long)H * HD * HD);
  sample_f16<<<ew_grid((long)TG * D), B256, 0, stream>>>(query, qs, TG, D, 8);
  sample_f16<<<ew_grid((long)TG * D), B256, 0, stream>>>(key,   ks, TG, D, 8);
  sample_f16<<<ew_grid((long)TG * D), B256, 0, stream>>>(value, vs, TG, D, 8);

  // 2) projections (WMMA GEMMs), all N=1024 slices of the packed 3D weights
  dim3 gP(T / 64, D / 128, 1), gPg(TG / 64, D / 128, 1);
  gemm_kernel<true, 4><<<gP, B256, 0, stream>>>(qf, D, 0, WqkvT_l + 0,    D3, 0, b_l + 0,    Qp_l, D, 0, D);
  gemm_kernel<true, 4><<<gP, B256, 0, stream>>>(kf, D, 0, WqkvT_l + D,    D3, 0, b_l + D,    Kp_l, D, 0, D);
  gemm_kernel<true, 4><<<gP, B256, 0, stream>>>(vf, D, 0, WqkvT_l + 2*D,  D3, 0, b_l + 2*D,  Vp_l, D, 0, D);
  gemm_kernel<true, 4><<<gP, B256, 0, stream>>>(qf, D, 0, WqkvT_lo + 0,   D3, 0, b_lo + 0,   Qp_lo, D, 0, D);
  gemm_kernel<true, 4><<<gP, B256, 0, stream>>>(kf, D, 0, WqkvT_lo + D,   D3, 0, b_lo + D,   Kp_lo, D, 0, D);
  gemm_kernel<true, 4><<<gP, B256, 0, stream>>>(vf, D, 0, WqkvT_lo + 2*D, D3, 0, b_lo + 2*D, Vp_lo, D, 0, D);
  gemm_kernel<true, 4><<<gPg, B256, 0, stream>>>(qs, D, 0, WqkvT_g + 0,   D3, 0, b_g + 0,    Qp_g, D, 0, D);
  gemm_kernel<true, 4><<<gPg, B256, 0, stream>>>(ks, D, 0, WqkvT_g + D,   D3, 0, b_g + D,    Kp_g, D, 0, D);
  gemm_kernel<true, 4><<<gPg, B256, 0, stream>>>(vs, D, 0, WqkvT_g + 2*D, D3, 0, b_g + 2*D,  Vp_g, D, 0, D);

  // 3) per-head quantum rotations: q_rot = qh @ R[h], k_rot = kh @ R[h]
  //    N=64 -> NFRAG=2 (block n-tile 64, exact fit, no guards)
  dim3 gR(T / 64, 1, H);
  gemm_kernel<true, 2><<<gR, B256, 0, stream>>>(qf, D, HD, Rf, HD, (long)HD * HD, nullptr, qrot, D, HD, HD);
  gemm_kernel<true, 2><<<gR, B256, 0, stream>>>(kf, D, HD, Rf, HD, (long)HD * HD, nullptr, krot, D, HD, HD);

  // 4) K transposes into [H][hd][T]
  head_transpose<<<ew_grid((long)H * HD * T), B256, 0, stream>>>(Kp_l, D, 0, KT_l,  H, HD, T);
  head_transpose<<<ew_grid((long)H * HD * T), B256, 0, stream>>>(kf,   D, 0, KT_sp, H, HD, T);
  head_transpose<<<ew_grid((long)H * HD * T), B256, 0, stream>>>(krot, D, 0, KT_qr, H, HD, T);
  head_transpose<<<ew_grid((long)H2 * HD2 * T), B256, 0, stream>>>(Kp_lo, D, 0, KT_lo, H2, HD2, T);
  head_transpose<<<ew_grid((long)H2 * HD2 * TG), B256, 0, stream>>>(Kp_g, D, 0, KT_g, H2, HD2, TG);

  // 5) attention variants (two-pass WMMA flash-softmax); grids tile exactly
  dim3 gA(H * (T / 16) / 8), gAl(H2 * (T / 16) / 8), gAg(H2 * (TG / 16) / 8);
  // linear
  attn_stats<64><<<gA, B256, 0, stream>>>(Qp_l, D, KT_l, sm1, sl1, H, T, T, ISC64, nullptr, 0);
  attn_apply<64><<<gA, B256, 0, stream>>>(Qp_l, D, KT_l, Vp_l, D, sm1, sl1, at_l_f, D,
                                          H, T, T, ISC64, nullptr, 0, combined, sprobs, 0);
  // sparse (raw heads, per-head 0/1 mask on scores)
  attn_stats<64><<<gA, B256, 0, stream>>>(qf, D, KT_sp, sm2, sl2, H, T, T, ISC64, spat, 1);
  attn_apply<64><<<gA, B256, 0, stream>>>(qf, D, KT_sp, vf, D, sm2, sl2, sparse_o, D,
                                          H, T, T, ISC64, spat, 1, combined, sprobs, 1);
  // quantum (rotated q/k, sigmoid gate on scores)
  attn_stats<64><<<gA, B256, 0, stream>>>(qrot, D, KT_qr, sm3, sl3, H, T, T, ISC64, entg, 2);
  attn_apply<64><<<gA, B256, 0, stream>>>(qrot, D, KT_qr, vf, D, sm3, sl3, quant_o, D,
                                          H, T, T, ISC64, entg, 2, combined, sprobs, 3);
  // local (8 heads, hd=128)
  attn_stats<128><<<gAl, B256, 0, stream>>>(Qp_lo, D, KT_lo, sm4, sl4, H2, T, T, ISC128, nullptr, 0);
  attn_apply<128><<<gAl, B256, 0, stream>>>(Qp_lo, D, KT_lo, Vp_lo, D, sm4, sl4, at_lo_f, D,
                                            H2, T, T, ISC128, nullptr, 0, nullptr, sprobs, 0);
  // global (subsampled, 8 heads, hd=128)
  attn_stats<128><<<gAg, B256, 0, stream>>>(Qp_g, D, KT_g, sm5, sl5, H2, TG, TG, ISC128, nullptr, 0);
  attn_apply<128><<<gAg, B256, 0, stream>>>(Qp_g, D, KT_g, Vp_g, D, sm5, sl5, at_g_f, D,
                                            H2, TG, TG, ISC128, nullptr, 0, nullptr, sprobs, 0);

  // 6) output projections
  f32_to_f16<<<ew_grid((long)T * D), B256, 0, stream>>>(at_l_f,  at_l_h,  (long)T * D);
  f32_to_f16<<<ew_grid((long)T * D), B256, 0, stream>>>(at_lo_f, at_lo_h, (long)T * D);
  f32_to_f16<<<ew_grid((long)TG * D), B256, 0, stream>>>(at_g_f, at_g_h, (long)TG * D);
  gemm_kernel<false, 4><<<gP, B256, 0, stream>>>(at_l_h,  D, 0, WoT_l,  D, 0, bo_l,  lin_o, D, 0, D);
  gemm_kernel<false, 4><<<gP, B256, 0, stream>>>(at_lo_h, D, 0, WoT_lo, D, 0, bo_lo, loc_o, D, 0, D);
  gemm_kernel<false, 4><<<gPg, B256, 0, stream>>>(at_g_h, D, 0, WoT_g,  D, 0, bo_g,  gsm,   D, 0, D);

  // 7) hierarchical fuse: concat(local, interp(global)) @ Wf.T + bf
  concat_interp<<<ew_grid((long)T * D), B256, 0, stream>>>(loc_o, gsm, catb, T, D, TG);
  gemm_kernel<false, 4><<<gP, B256, 0, stream>>>(catb, D2, 0, WfT, D, 0, bf, hier_o, D, 0, D2);

  // 8) strategy-weighted combination of the four outputs
  final_combine<<<ew_grid((long)T * D), B256, 0, stream>>>(lin_o, sparse_o, hier_o, quant_o,
                                                           sprobs, weighted, (long)T * D);
}
